// MultiHeadAttention_75033078661863
// MI455X (gfx1250) — matrix operand order repaired and checked by reading, not compiled, not run on hardware
//
#include <hip/hip_runtime.h>
#include <hip/hip_bf16.h>

// MI455X (gfx1250) fused MHA, all matmuls on v_wmma_f32_16x16x32_f16 (f32 accum).
//  kernel 1: Q^T/K^T = W x^T  (C rows = d -> b128 stores into [s][d]);
//            V       = x W^T  (C rows = t -> b128 stores into transposed [d][t])
//  kernel 2: 256-thread blocks (8 waves, one (b,h), 256 queries). K / V^T
//            32-key blocks staged into LDS with global_load_async_to_lds_b128,
//            double-buffered on ASYNCcnt (s_wait_asynccnt) + block barriers ->
//            8x less global traffic than per-wave fragment loads.
//            S^T = K Q^T (softmax stats are per-lane scalars, one shfl each);
//            O^T = V^T P^T (P^T via per-wave LDS b128 round-trip).
// ~28 GFLOP total; 37.7 MB f16 Q/K/V workspace stays resident in the 192 MB L2.

#define B_  8
#define S_  1024
#define D_  768
#define H_  12
#define DH_ 64

typedef _Float16 half_t;
typedef half_t v8h  __attribute__((ext_vector_type(8)));
typedef half_t v16h __attribute__((ext_vector_type(16)));
typedef float  v8f  __attribute__((ext_vector_type(8)));

// A/B fragment element order: [8 contiguous f16][next 8 contiguous f16]
static __device__ inline v16h cat8(v8h a, v8h b) {
    v16h r;
#pragma unroll
    for (int j = 0; j < 8; ++j) { r[j] = a[j]; r[j + 8] = b[j]; }
    return r;
}

// load 8 consecutive f32, convert to 8 f16
static __device__ inline v8h cvt8f(const float* __restrict__ p) {
    float4 f0 = *(const float4*)p;
    float4 f1 = *(const float4*)(p + 4);
    v8h r;
    r[0] = (half_t)f0.x; r[1] = (half_t)f0.y; r[2] = (half_t)f0.z; r[3] = (half_t)f0.w;
    r[4] = (half_t)f1.x; r[5] = (half_t)f1.y; r[6] = (half_t)f1.z; r[7] = (half_t)f1.w;
    return r;
}

// async 16B global -> LDS copy (gfx1250 VGLOBAL async op, tracked by ASYNCcnt)
static __device__ inline void async_copy_b128(unsigned int lds_off, const half_t* g) {
    asm volatile("global_load_async_to_lds_b128 %0, %1, off"
                 :: "v"(lds_off), "v"((unsigned long long)(uintptr_t)g)
                 : "memory");
}

// ---------------------------------------------------------------------------
// Kernel 1: QKV projection, one wave = one 16-row tile of one (b,h).
// ---------------------------------------------------------------------------
__global__ __launch_bounds__(128) void qkv_proj_kernel(
    const float* __restrict__ x,
    const float* __restrict__ Wq, const float* __restrict__ bq,
    const float* __restrict__ Wk, const float* __restrict__ bk,
    const float* __restrict__ Wv, const float* __restrict__ bv,
    half_t* __restrict__ qws, half_t* __restrict__ kws, half_t* __restrict__ vtws)
{
    const int lane = threadIdx.x & 31;
    const int wave = threadIdx.x >> 5;
    const int job  = blockIdx.x * 4 + wave;   // B*H*(S/16) = 6144 jobs
    const int bh = job >> 6;
    const int s0 = (job & 63) << 4;
    const int h  = bh % H_;
    const int b  = bh / H_;

    const int nl = lane & 15;
    const int hi = lane >> 4;

    const float* xrow = x + ((size_t)(b * S_ + s0 + nl)) * D_ + h * DH_;
    // x as A (rows = s, k = e)
    const v16h ax0 = cat8(cvt8f(xrow + 8 * hi),      cvt8f(xrow + 16 + 8 * hi));
    const v16h ax1 = cat8(cvt8f(xrow + 32 + 8 * hi), cvt8f(xrow + 48 + 8 * hi));
    // x^T as B (k = e, n = s)
    const v16h bx0 = cat8(cvt8f(xrow + 8 * hi),      cvt8f(xrow + 16 + 8 * hi));
    const v16h bx1 = cat8(cvt8f(xrow + 32 + 8 * hi), cvt8f(xrow + 48 + 8 * hi));

    // Q^T / K^T tile: A = W rows (d), B = x^T; C rows m = d -> one b128 store
    auto projT = [&](const float* __restrict__ W, const float* __restrict__ bias,
                     half_t* __restrict__ dst, int n0, float scale) {
        const float* wrow = W + ((size_t)h * DH_ + n0 + nl) * DH_;
        v16h a0 = cat8(cvt8f(wrow + 8 * hi),      cvt8f(wrow + 16 + 8 * hi));
        v16h a1 = cat8(cvt8f(wrow + 32 + 8 * hi), cvt8f(wrow + 48 + 8 * hi));
        v8f c = {};
        c = __builtin_amdgcn_wmma_f32_16x16x32_f16(false, a0, false, bx0, (short)0, c, false, false);
        c = __builtin_amdgcn_wmma_f32_16x16x32_f16(false, a1, false, bx1, (short)0, c, false, false);
        const float* bp = bias + h * DH_ + n0 + 8 * hi;   // bias over m = d
        float4 bb0 = *(const float4*)bp;
        float4 bb1 = *(const float4*)(bp + 4);
        const float bb[8] = {bb0.x, bb0.y, bb0.z, bb0.w, bb1.x, bb1.y, bb1.z, bb1.w};
        v8h pk;
#pragma unroll
        for (int i = 0; i < 8; ++i) pk[i] = (half_t)((c[i] + bb[i]) * scale);
        *(v8h*)(dst + ((size_t)bh * S_ + s0 + nl) * DH_ + n0 + 8 * hi) = pk;
    };

#pragma unroll
    for (int t = 0; t < 4; ++t) {
        const int n0 = t * 16;
        projT(Wq, bq, qws, n0, 0.125f);   // fold 1/sqrt(DH) into Q
        projT(Wk, bk, kws, n0, 1.0f);

        // V tile: A = x (rows s), B = Wv^T (k = e, n = d); C rows m = t
        const float* wrow = Wv + ((size_t)h * DH_ + n0 + nl) * DH_;
        v16h b0 = cat8(cvt8f(wrow + 8 * hi),      cvt8f(wrow + 16 + 8 * hi));
        v16h b1 = cat8(cvt8f(wrow + 32 + 8 * hi), cvt8f(wrow + 48 + 8 * hi));
        v8f c = {};
        c = __builtin_amdgcn_wmma_f32_16x16x32_f16(false, ax0, false, b0, (short)0, c, false, false);
        c = __builtin_amdgcn_wmma_f32_16x16x32_f16(false, ax1, false, b1, (short)0, c, false, false);
        const float bb = bv[h * DH_ + n0 + nl];
        v8h pk;
#pragma unroll
        for (int i = 0; i < 8; ++i) pk[i] = (half_t)(c[i] + bb);
        *(v8h*)(vtws + ((size_t)bh * DH_ + n0 + nl) * S_ + s0 + 8 * hi) = pk;
    }
}

// ---------------------------------------------------------------------------
// Kernel 2: flash attention. Block = 8 waves = 256 queries of one (b,h).
// K/V^T 32-key blocks are staged into LDS with async copies (double-buffered).
// ---------------------------------------------------------------------------
#define KPAD 72   // 32 x 64-half K tile, padded row stride (36 dwords, conflict-free)
#define VPAD 40   // 64 x 32-half V^T tile, padded row stride (20 dwords)
#define NIT  (S_ / 32)

__global__ __launch_bounds__(256) void attn_kernel(
    const half_t* __restrict__ qws, const half_t* __restrict__ kws,
    const half_t* __restrict__ vtws, float* __restrict__ out)
{
    __shared__ half_t Kbuf[2][32][KPAD];     //  9216 B
    __shared__ half_t Vbuf[2][64][VPAD];     // 10240 B
    __shared__ half_t Plds[8][2][16][40];    // 20480 B (per-wave P^T round-trip)

    const int tid  = threadIdx.x;
    const int lane = tid & 31;
    const int wave = tid >> 5;
    const int job  = blockIdx.x * 8 + wave;   // B*H*(S/32) = 3072 jobs
    const int bh = job >> 5;                  // all 8 waves share bh
    const int q0 = (job & 31) << 5;
    const int h  = bh % H_;
    const int b  = bh / H_;

    const int nl = lane & 15;
    const int hi = lane >> 4;

    const half_t* kbase = kws  + (size_t)bh * S_ * DH_;
    const half_t* vbase = vtws + (size_t)bh * DH_ * S_;

    // one 16B K chunk + one 16B V chunk per thread per block-copy (2 per wave)
    const int krow = tid >> 3, kcq = tid & 7;   // K: 32 rows x 8 chunks
    const int vrw  = tid >> 2, vcq = tid & 3;   // V: 64 rows x 4 chunks
    auto issue_copy = [&](int t0, int buf) {
        async_copy_b128((unsigned int)(uintptr_t)&Kbuf[buf][krow][kcq * 8],
                        kbase + (size_t)(t0 + krow) * DH_ + kcq * 8);
        async_copy_b128((unsigned int)(uintptr_t)&Vbuf[buf][vrw][vcq * 8],
                        vbase + (size_t)vrw * S_ + t0 + vcq * 8);
    };

    // Q^T B-fragments (k = d, n = q), per-lane contiguous loads
    v16h bqf[2][2];
#pragma unroll
    for (int qt = 0; qt < 2; ++qt) {
        const half_t* qrow = qws + ((size_t)bh * S_ + q0 + qt * 16 + nl) * DH_;
#pragma unroll
        for (int kc = 0; kc < 2; ++kc)
            bqf[qt][kc] = cat8(*(const v8h*)(qrow + kc * 32 + 8 * hi),
                               *(const v8h*)(qrow + kc * 32 + 16 + 8 * hi));
    }

    v8f   O[2][4];                       // O^T tiles: rows = d, cols = q
    float mq[2] = {-1e30f, -1e30f};
    float lq[2] = {0.0f, 0.0f};
#pragma unroll
    for (int qt = 0; qt < 2; ++qt)
#pragma unroll
        for (int dv = 0; dv < 4; ++dv)
#pragma unroll
            for (int i = 0; i < 8; ++i) O[qt][dv][i] = 0.0f;

    issue_copy(0, 0);                    // prologue: stage first key block

    for (int it = 0; it < NIT; ++it) {
        const int cur = it & 1;
        if (it + 1 < NIT) {
            issue_copy((it + 1) * 32, cur ^ 1);          // prefetch next block
            asm volatile("s_wait_asynccnt 0x2" ::: "memory");  // cur copies done
        } else {
            asm volatile("s_wait_asynccnt 0x0" ::: "memory");
        }
        __syncthreads();                 // cur buffer visible block-wide

        // K A-fragments (rows = keys, k = d) from LDS
        v16h ak[2][2];
#pragma unroll
        for (int kt = 0; kt < 2; ++kt) {
            const half_t* kr = &Kbuf[cur][kt * 16 + nl][0];
            ak[kt][0] = cat8(*(const v8h*)(kr + 8 * hi),      *(const v8h*)(kr + 16 + 8 * hi));
            ak[kt][1] = cat8(*(const v8h*)(kr + 32 + 8 * hi), *(const v8h*)(kr + 48 + 8 * hi));
        }

#pragma unroll
        for (int qt = 0; qt < 2; ++qt) {
            // S^T tiles: 32 keys x 16 queries (Q pre-scaled by 1/8)
            v8f c0 = {}, c1 = {};
            c0 = __builtin_amdgcn_wmma_f32_16x16x32_f16(false, ak[0][0], false, bqf[qt][0], (short)0, c0, false, false);
            c0 = __builtin_amdgcn_wmma_f32_16x16x32_f16(false, ak[0][1], false, bqf[qt][1], (short)0, c0, false, false);
            c1 = __builtin_amdgcn_wmma_f32_16x16x32_f16(false, ak[1][0], false, bqf[qt][0], (short)0, c1, false, false);
            c1 = __builtin_amdgcn_wmma_f32_16x16x32_f16(false, ak[1][1], false, bqf[qt][1], (short)0, c1, false, false);

            // softmax over keys: in-lane reduce + one cross-half shuffle
            float tm = fmaxf(c0[0], c1[0]);
#pragma unroll
            for (int i = 1; i < 8; ++i) tm = fmaxf(tm, fmaxf(c0[i], c1[i]));
            tm = fmaxf(tm, __shfl_xor(tm, 16, 32));
            const float nm   = fmaxf(mq[qt], tm);
            const float corr = __expf(mq[qt] - nm);
            mq[qt] = nm;

            float rs = 0.0f;
#pragma unroll
            for (int i = 0; i < 8; ++i) {
                c0[i] = __expf(c0[i] - nm);
                c1[i] = __expf(c1[i] - nm);
                rs += c0[i] + c1[i];
            }
            rs += __shfl_xor(rs, 16, 32);
            lq[qt] = lq[qt] * corr + rs;

#pragma unroll
            for (int dv = 0; dv < 4; ++dv)
#pragma unroll
                for (int i = 0; i < 8; ++i) O[qt][dv][i] *= corr;

            // P^T -> LDS as [query][key]: one b128 store per tile
            v8h p0, p1;
#pragma unroll
            for (int i = 0; i < 8; ++i) { p0[i] = (half_t)c0[i]; p1[i] = (half_t)c1[i]; }
            *(v8h*)&Plds[wave][qt][nl][8 * hi]      = p0;
            *(v8h*)&Plds[wave][qt][nl][16 + 8 * hi] = p1;
        }

        // P^T B-fragments (k = keys, n = q)
        v16h bp[2];
#pragma unroll
        for (int qt = 0; qt < 2; ++qt) {
            const half_t* prow = &Plds[wave][qt][nl][0];
            bp[qt] = cat8(*(const v8h*)(prow + 8 * hi), *(const v8h*)(prow + 16 + 8 * hi));
        }

        // O^T += V^T P^T : V^T A-fragments from LDS
#pragma unroll
        for (int dv = 0; dv < 4; ++dv) {
            const half_t* vr = &Vbuf[cur][dv * 16 + nl][0];
            v16h av = cat8(*(const v8h*)(vr + 8 * hi), *(const v8h*)(vr + 16 + 8 * hi));
            O[0][dv] = __builtin_amdgcn_wmma_f32_16x16x32_f16(false, av, false, bp[0], (short)0, O[0][dv], false, false);
            O[1][dv] = __builtin_amdgcn_wmma_f32_16x16x32_f16(false, av, false, bp[1], (short)0, O[1][dv], false, false);
        }

        __syncthreads();                 // all reads of cur done -> reusable
    }

    // epilogue: O^T rows are d -> contiguous f32 stores (2 x b128 per tile)
#pragma unroll
    for (int qt = 0; qt < 2; ++qt) {
        const float inv = 1.0f / lq[qt];
        const int q = q0 + qt * 16 + nl;
#pragma unroll
        for (int dv = 0; dv < 4; ++dv) {
            float* op = out + ((size_t)(b * S_ + q)) * D_ + h * DH_ + dv * 16 + 8 * hi;
            float4 o0 = {O[qt][dv][0] * inv, O[qt][dv][1] * inv, O[qt][dv][2] * inv, O[qt][dv][3] * inv};
            float4 o1 = {O[qt][dv][4] * inv, O[qt][dv][5] * inv, O[qt][dv][6] * inv, O[qt][dv][7] * inv};
            *(float4*)op       = o0;
            *(float4*)(op + 4) = o1;
        }
    }
}

extern "C" void kernel_launch(void* const* d_in, const int* in_sizes, int n_in,
                              void* d_out, int out_size, void* d_ws, size_t ws_size,
                              hipStream_t stream) {
    const float* x  = (const float*)d_in[0];
    const float* Wq = (const float*)d_in[1];
    const float* bq = (const float*)d_in[2];
    const float* Wk = (const float*)d_in[3];
    const float* bk = (const float*)d_in[4];
    const float* Wv = (const float*)d_in[5];
    const float* bv = (const float*)d_in[6];
    float* out = (float*)d_out;

    const size_t N = (size_t)B_ * H_ * S_ * DH_;   // 6,291,456 elements per tensor
    half_t* qws = (half_t*)d_ws;                   // [bh][s][d]
    half_t* kws = qws + N;                         // [bh][s][d]
    half_t* vt  = kws + N;                         // [bh][d][t]

    qkv_proj_kernel<<<1536, 128, 0, stream>>>(x, Wq, bq, Wk, bk, Wv, bv, qws, kws, vt);
    attn_kernel<<<384, 256, 0, stream>>>(qws, kws, vt, out);
}
